// PTFactorNoisePredictor_73280732004907
// MI455X (gfx1250) — compile-verified, matrix operand order checked by reading, not run on hardware
//
#include <hip/hip_runtime.h>
#include <hip/hip_bf16.h>
#include <math.h>

#define B_ 8
#define C_ 32
#define P_ 256
#define D_ 256
#define G_ 512
#define H_ 8
#define R_ 32

typedef __attribute__((ext_vector_type(16))) __bf16 v16bf;
typedef __attribute__((ext_vector_type(8)))  float  v8f;

union FragU { uint4 q[2]; v16bf v; };

// Fragment load for 16x16x32 bf16 WMMA A/B operands from a row-major,
// K-contiguous matrix. ISA layout (CDNA5 ISA):
//   lanes 0-15 : row = tile_row + lane,      K = k+0..7  and k+16..23
//   lanes 16-31: row = tile_row + lane-16,   K = k+8..15 and k+24..31
// -> two contiguous 16B loads per lane.
__device__ __forceinline__ v16bf load_frag(const __bf16* base, int row, int ld,
                                           int k, int lane) {
  const __bf16* p = base + (size_t)row * (size_t)ld + (size_t)(k + ((lane >= 16) ? 8 : 0));
  FragU f;
  f.q[0] = *reinterpret_cast<const uint4*>(p);
  f.q[1] = *reinterpret_cast<const uint4*>(p + 16);
  return f.v;
}

__device__ __forceinline__ v8f wmma_bf16(v16bf a, v16bf b, v8f c) {
  return __builtin_amdgcn_wmma_f32_16x16x32_bf16(false, a, false, b, (short)0, c,
                                                 false, false);
}

// ---------------------------------------------------------------- elementwise
__global__ void k_f32_to_bf16(const float* __restrict__ in, __bf16* __restrict__ out,
                              long long n) {
  long long i = (long long)blockIdx.x * blockDim.x + threadIdx.x;
  if (i < n) out[i] = (__bf16)in[i];
}

__global__ void k_zero_f32(float* __restrict__ p, long long n) {
  long long i = (long long)blockIdx.x * blockDim.x + threadIdx.x;
  if (i < n) p[i] = 0.0f;
}

// qznT[b,c,d,p] = qzn[b,c,p,d]
__global__ void k_tr_pd(const __bf16* __restrict__ in, __bf16* __restrict__ out,
                        long long n) {
  long long i = (long long)blockIdx.x * blockDim.x + threadIdx.x;
  if (i >= n) return;
  int p  = (int)(i & (P_ - 1));
  int d  = (int)((i >> 8) & (D_ - 1));
  long long bc = i >> 16;
  out[i] = in[(bc << 16) + ((long long)p << 8) + d];
}

// qzCT[b,p,d,c] = qzn[b,c,p,d]
__global__ void k_tr_ct(const __bf16* __restrict__ in, __bf16* __restrict__ out,
                        long long n) {
  long long i = (long long)blockIdx.x * blockDim.x + threadIdx.x;
  if (i >= n) return;
  int c = (int)(i & (C_ - 1));
  int d = (int)((i >> 5) & (D_ - 1));
  int p = (int)((i >> 13) & (P_ - 1));
  int b = (int)(i >> 21);
  out[i] = in[((((long long)b * C_ + c) * P_ + p) << 8) + d];
}

// topicT[b,d,g] = (bf16)topic[b,g,d]
__global__ void k_tr_topic(const float* __restrict__ in, __bf16* __restrict__ out,
                           long long n) {
  long long i = (long long)blockIdx.x * blockDim.x + threadIdx.x;
  if (i >= n) return;
  int g = (int)(i & (G_ - 1));
  int d = (int)((i >> 9) & (D_ - 1));
  int b = (int)(i >> 17);
  out[i] = (__bf16)in[(((long long)b * G_ + g) << 8) + d];
}

// ---------------------------------------------------------------- layernorm
// one block (256 threads) per row of D=256
__global__ void k_layernorm(const float* __restrict__ x, const float* __restrict__ sc,
                            const float* __restrict__ bi, __bf16* __restrict__ out) {
  int row = blockIdx.x;
  int t = threadIdx.x;
  __shared__ float red[8];
  size_t base = (size_t)row * D_;
  float v = x[base + t];
  float s = v;
  for (int o = 16; o > 0; o >>= 1) s += __shfl_down(s, o);
  if ((t & 31) == 0) red[t >> 5] = s;
  __syncthreads();
  float mean = 0.f;
  for (int i = 0; i < 8; ++i) mean += red[i];
  mean *= (1.0f / D_);
  __syncthreads();
  float d = v - mean;
  float s2 = d * d;
  for (int o = 16; o > 0; o >>= 1) s2 += __shfl_down(s2, o);
  if ((t & 31) == 0) red[t >> 5] = s2;
  __syncthreads();
  float var = 0.f;
  for (int i = 0; i < 8; ++i) var += red[i];
  var *= (1.0f / D_);
  float y = d * rsqrtf(var + 1e-5f) * sc[t] + bi[t];
  out[base + t] = (__bf16)y;
}

// ---------------------------------------------------------------- generic GEMM
// C[bat] = epilogue( A[bat](MxK) @ B[bat](NxK)^T ), bf16 operands, f32 accum.
// 128 threads = 4 waves in 2x2; block tile 64x64; per-wave 32x32 (4 WMMAs/kstep).
enum {
  EPI_STORE_BF16 = 0,
  EPI_STORE_F32  = 1,
  EPI_ACC_F32    = 2,
  EPI_RELU_RS    = 3,   // relu, store bf16, atomic rowsum
  EPI_TOPIC2     = 4,   // accumulate v / max(rowsum,1e-6) into f32
  EPI_GATE       = 5,   // sigmoid gate blend -> f32 out
  EPI_GELU       = 6,   // exact gelu(v+bias) -> bf16 out
  EPI_ADDOUT     = 7    // out += v + bias
};

struct GemmP {
  const __bf16* A;
  const __bf16* B;
  int M, N, K;
  int lda, ldb, ldc;
  long long asr, bsr;           // batch strides (elements)
  int bdiv; long long cs0, cs1; // cbase = (bat/bdiv)*cs0 + (bat%bdiv)*cs1
  const float* bias;
  float* outF;
  __bf16* outB;
  const float* aux0;            // qz
  const float* aux1;            // unary
  const float* aux2;            // msum
  float* rowsum;
};

template <int EPI>
__global__ __launch_bounds__(128) void k_gemm(GemmP g) {
  const int lane = threadIdx.x & 31;
  const int wave = threadIdx.x >> 5;
  const int wm = wave >> 1, wn = wave & 1;
  const int row0 = blockIdx.y * 64 + wm * 32;
  const int col0 = blockIdx.x * 64 + wn * 32;
  const int bat = blockIdx.z;

  const __bf16* A = g.A + (long long)bat * g.asr;
  const __bf16* Bm = g.B + (long long)bat * g.bsr;
  const size_t cbase = (size_t)((long long)(bat / g.bdiv) * g.cs0 +
                                (long long)(bat % g.bdiv) * g.cs1);

  int rA0 = row0 + (lane & 15);       if (rA0 > g.M - 1) rA0 = g.M - 1;
  int rA1 = row0 + 16 + (lane & 15);  if (rA1 > g.M - 1) rA1 = g.M - 1;
  int cB0 = col0 + (lane & 15);       if (cB0 > g.N - 1) cB0 = g.N - 1;
  int cB1 = col0 + 16 + (lane & 15);  if (cB1 > g.N - 1) cB1 = g.N - 1;

  v8f acc00 = {}, acc01 = {}, acc10 = {}, acc11 = {};
  for (int k = 0; k < g.K; k += 32) {
    v16bf a0 = load_frag(A, rA0, g.lda, k, lane);
    v16bf a1 = load_frag(A, rA1, g.lda, k, lane);
    v16bf b0 = load_frag(Bm, cB0, g.ldb, k, lane);
    v16bf b1 = load_frag(Bm, cB1, g.ldb, k, lane);
    acc00 = wmma_bf16(a0, b0, acc00);
    acc01 = wmma_bf16(a0, b1, acc01);
    acc10 = wmma_bf16(a1, b0, acc10);
    acc11 = wmma_bf16(a1, b1, acc11);
  }

  const int rb = (lane >= 16) ? 8 : 0;
#pragma unroll
  for (int mt = 0; mt < 2; ++mt) {
#pragma unroll
    for (int nt = 0; nt < 2; ++nt) {
      const v8f& ac = (mt == 0) ? (nt == 0 ? acc00 : acc01)
                                : (nt == 0 ? acc10 : acc11);
      int colg = col0 + nt * 16 + (lane & 15);
#pragma unroll
      for (int r = 0; r < 8; ++r) {
        int rowg = row0 + mt * 16 + rb + r;
        if (rowg < g.M && colg < g.N) {
          float v = ac[r];
          size_t ci = cbase + (size_t)rowg * (size_t)g.ldc + (size_t)colg;
          if (EPI == EPI_STORE_BF16) {
            g.outB[ci] = (__bf16)v;
          } else if (EPI == EPI_STORE_F32) {
            g.outF[ci] = v;
          } else if (EPI == EPI_ACC_F32) {
            g.outF[ci] += v;
          } else if (EPI == EPI_RELU_RS) {
            v = fmaxf(v, 0.0f);
            g.outB[ci] = (__bf16)v;
            atomicAdd(&g.rowsum[(size_t)bat * (size_t)g.M + rowg], v);
          } else if (EPI == EPI_TOPIC2) {
            float s = g.rowsum[(size_t)bat * (size_t)g.M + rowg];
            g.outF[ci] += v / fmaxf(s, 1e-6f);
          } else if (EPI == EPI_GATE) {
            float gt = 1.0f / (1.0f + __expf(-(v + g.bias[colg])));
            g.outF[ci] = gt * g.aux0[ci] + (1.0f - gt) * (g.aux1[ci] + g.aux2[ci]);
          } else if (EPI == EPI_GELU) {
            float x = v + g.bias[colg];
            g.outB[ci] = (__bf16)(0.5f * x * (1.0f + erff(x * 0.70710678118654752f)));
          } else if (EPI == EPI_ADDOUT) {
            g.outF[ci] += v + g.bias[colg];
          }
        }
      }
    }
  }
}

// ---------------------------------------------------------------- time attention
// per (b,c): 16-row strip of p; for each head build logits strip with WMMA,
// softmax over q (256) in LDS, accumulate head-averaged probabilities.
__global__ __launch_bounds__(128) void k_attn_time(const __bf16* __restrict__ Ut,
                                                   const __bf16* __restrict__ Vt,
                                                   __bf16* __restrict__ attnOut) {
  const int pt = blockIdx.x;   // 0..15 (p strip)
  const int bc = blockIdx.y;   // 0..B*C-1
  const int tid = threadIdx.x; // 0..127
  const int wave = tid >> 5, lane = tid & 31;

  __shared__ float S[16][256];
  __shared__ float Pav[16][256];
  __shared__ float red[16][8];

  const __bf16* ub = Ut + (size_t)bc * P_ * D_;
  const __bf16* vb = Vt + (size_t)bc * P_ * D_;
  const float invs = 0.17677669529663687f; // 1/sqrt(R=32)

  for (int h = 0; h < H_; ++h) {
    const int hc = h * R_;
    v16bf a = load_frag(ub, pt * 16 + (lane & 15), D_, hc, lane);
    const int rb = (lane >= 16) ? 8 : 0;
#pragma unroll
    for (int t = 0; t < 4; ++t) {
      int qt = wave * 4 + t;
      v16bf bf = load_frag(vb, qt * 16 + (lane & 15), D_, hc, lane);
      v8f c = {};
      c = wmma_bf16(a, bf, c);
      int col = qt * 16 + (lane & 15);
#pragma unroll
      for (int r = 0; r < 8; ++r) S[rb + r][col] = c[r] * invs;
    }
    __syncthreads();

    // softmax rows (8 threads per row, 32 cols each)
    const int row = tid >> 3, seg = tid & 7;
    float mx = -1e30f;
    for (int j = 0; j < 32; ++j) mx = fmaxf(mx, S[row][seg * 32 + j]);
    red[row][seg] = mx;
    __syncthreads();
    mx = red[row][0];
    for (int j = 1; j < 8; ++j) mx = fmaxf(mx, red[row][j]);
    __syncthreads();
    float e[32];
    float sm = 0.0f;
    for (int j = 0; j < 32; ++j) {
      float v = __expf(S[row][seg * 32 + j] - mx);
      e[j] = v;
      sm += v;
    }
    red[row][seg] = sm;
    __syncthreads();
    sm = 0.0f;
    for (int j = 0; j < 8; ++j) sm += red[row][j];
    float inv = 1.0f / sm;
    for (int j = 0; j < 32; ++j) {
      float pv = e[j] * inv;
      if (h == 0) Pav[row][seg * 32 + j] = pv;
      else        Pav[row][seg * 32 + j] += pv;
    }
    __syncthreads();
  }

  __bf16* outp = attnOut + (size_t)bc * P_ * P_ + (size_t)pt * 16 * P_;
  for (int i = tid; i < 16 * 256; i += 128) {
    int r = i >> 8, cn = i & 255;
    outp[(size_t)r * P_ + cn] = (__bf16)(Pav[r][cn] * (1.0f / H_));
  }
}

// ---------------------------------------------------------------- channel attention
// per (b,p): 32x32 logits per head via 4 WMMAs, softmax, head-average.
__global__ __launch_bounds__(32) void k_attn_channel(const __bf16* __restrict__ Uc,
                                                     const __bf16* __restrict__ Vc,
                                                     __bf16* __restrict__ attnOut) {
  const int bp = blockIdx.x;          // b*P + p
  const int b = bp >> 8, p = bp & 255;
  const int lane = threadIdx.x;

  __shared__ float S[32][33];
  __shared__ float Pav[32][32];

  const size_t rowstr = (size_t)P_ * D_; // stride between c-rows
  const __bf16* ub = Uc + ((size_t)b * C_ * P_ + p) * D_;
  const __bf16* vb = Vc + ((size_t)b * C_ * P_ + p) * D_;
  const float invs = 0.17677669529663687f;
  const int rb = (lane >= 16) ? 8 : 0;

  for (int h = 0; h < H_; ++h) {
    const int hc = h * R_;
    v16bf a0 = load_frag(ub, (lane & 15), (int)rowstr, hc, lane);
    v16bf a1 = load_frag(ub, 16 + (lane & 15), (int)rowstr, hc, lane);
    v16bf b0 = load_frag(vb, (lane & 15), (int)rowstr, hc, lane);
    v16bf b1 = load_frag(vb, 16 + (lane & 15), (int)rowstr, hc, lane);
    v8f c00 = {}, c01 = {}, c10 = {}, c11 = {};
    c00 = wmma_bf16(a0, b0, c00);
    c01 = wmma_bf16(a0, b1, c01);
    c10 = wmma_bf16(a1, b0, c10);
    c11 = wmma_bf16(a1, b1, c11);
    int cn = lane & 15;
#pragma unroll
    for (int r = 0; r < 8; ++r) {
      S[rb + r][cn]           = c00[r] * invs;
      S[rb + r][16 + cn]      = c01[r] * invs;
      S[16 + rb + r][cn]      = c10[r] * invs;
      S[16 + rb + r][16 + cn] = c11[r] * invs;
    }
    __syncthreads();
    // one thread per row
    float mx = -1e30f;
    for (int j = 0; j < 32; ++j) mx = fmaxf(mx, S[lane][j]);
    float sm = 0.0f;
    float e[32];
    for (int j = 0; j < 32; ++j) { e[j] = __expf(S[lane][j] - mx); sm += e[j]; }
    float inv = 1.0f / sm;
    for (int j = 0; j < 32; ++j) {
      float pv = e[j] * inv;
      if (h == 0) Pav[lane][j] = pv;
      else        Pav[lane][j] += pv;
    }
    __syncthreads();
  }

  __bf16* outp = attnOut + (size_t)bp * C_ * C_;
  for (int i = lane; i < 32 * 32; i += 32) {
    int r = i >> 5, cn = i & 31;
    outp[i] = (__bf16)(Pav[r][cn] * (1.0f / H_));
  }
}

// ---------------------------------------------------------------- launch
extern "C" void kernel_launch(void* const* d_in, const int* in_sizes, int n_in,
                              void* d_out, int out_size, void* d_ws, size_t ws_size,
                              hipStream_t stream) {
  const float* qz       = (const float*)d_in[0];
  const float* unary    = (const float*)d_in[1];
  const float* time_u   = (const float*)d_in[2];
  const float* time_v   = (const float*)d_in[3];
  const float* chan_u   = (const float*)d_in[4];
  const float* chan_v   = (const float*)d_in[5];
  const float* topic    = (const float*)d_in[6];
  const float* nsc      = (const float*)d_in[7];
  const float* nbi      = (const float*)d_in[8];
  const float* gate_w   = (const float*)d_in[9];
  const float* gate_b   = (const float*)d_in[10];
  const float* w1       = (const float*)d_in[11];
  const float* b1       = (const float*)d_in[12];
  const float* w2       = (const float*)d_in[13];
  const float* b2       = (const float*)d_in[14];
  float* out = (float*)d_out;
  char* ws = (char*)d_ws;

  const size_t MB = 1ull << 20;
  __bf16* qzn      = (__bf16*)(ws + 0);        // 32MB  LN(qz) bf16
  __bf16* qznt     = (__bf16*)(ws + 32 * MB);  // 32MB  [B,C,D,P]   (later: hbuf)
  __bf16* qzct     = (__bf16*)(ws + 64 * MB);  // 32MB  [B,P,D,C]   (later: tbuf)
  __bf16* proj     = (__bf16*)(ws + 96 * MB);  // 128MB Ut,Vt,Uc,Vc (later: qg)
  __bf16* attnT    = (__bf16*)(ws + 224 * MB); // 32MB  [B,C,P,P]   (later: qzbf)
  __bf16* attnC    = (__bf16*)(ws + 256 * MB); // 4MB   [B,P,C,C]
  float*  msum     = (float*)(ws + 261 * MB);  // 64MB  m_t+m_c+m_g fp32
  float*  rowsum   = (float*)(ws + 325 * MB);  // 256KB
  __bf16* wtu      = (__bf16*)(ws + 326 * MB); // 1MB each
  __bf16* wtv      = (__bf16*)(ws + 327 * MB);
  __bf16* wcu      = (__bf16*)(ws + 328 * MB);
  __bf16* wcv      = (__bf16*)(ws + 329 * MB);
  __bf16* topic_bf = (__bf16*)(ws + 330 * MB); // 2MB [B,G,D]
  __bf16* topicT   = (__bf16*)(ws + 332 * MB); // 2MB [B,D,G]
  __bf16* gwb      = (__bf16*)(ws + 334 * MB); // 128KB
  __bf16* w1b      = (__bf16*)(ws + 335 * MB);
  __bf16* w2b      = (__bf16*)(ws + 336 * MB);
  // lifetime-safe aliases
  __bf16* qg   = proj;   // topic activations (64MB), proj dead by then
  __bf16* qzbf = attnT;  // raw qz bf16 for gate GEMM, attnT dead by then
  __bf16* hbuf = qznt;   // LN(qz2) bf16
  __bf16* tbuf = qzct;   // gelu output bf16

  auto cdiv = [](long long a, long long b) { return (unsigned)((a + b - 1) / b); };
  const long long nq = (long long)B_ * C_ * P_ * D_; // 16,777,216
  const long long nw = (long long)B_ * D_ * D_;      // 524,288
  const long long ntb = (long long)B_ * G_ * D_;     // 1,048,576
  const long long nrows = (long long)B_ * C_ * P_;   // 65,536

  // 1. weight conversions
  k_f32_to_bf16<<<cdiv(nw, 256), 256, 0, stream>>>(time_u, wtu, nw);
  k_f32_to_bf16<<<cdiv(nw, 256), 256, 0, stream>>>(time_v, wtv, nw);
  k_f32_to_bf16<<<cdiv(nw, 256), 256, 0, stream>>>(chan_u, wcu, nw);
  k_f32_to_bf16<<<cdiv(nw, 256), 256, 0, stream>>>(chan_v, wcv, nw);
  k_f32_to_bf16<<<cdiv(65536, 256), 256, 0, stream>>>(gate_w, gwb, 65536);
  k_f32_to_bf16<<<cdiv(65536, 256), 256, 0, stream>>>(w1, w1b, 65536);
  k_f32_to_bf16<<<cdiv(65536, 256), 256, 0, stream>>>(w2, w2b, 65536);
  k_f32_to_bf16<<<cdiv(ntb, 256), 256, 0, stream>>>(topic, topic_bf, ntb);
  k_tr_topic<<<cdiv(ntb, 256), 256, 0, stream>>>(topic, topicT, ntb);
  k_zero_f32<<<cdiv(nrows, 256), 256, 0, stream>>>(rowsum, nrows);

  // 2. layernorm + transposed copies
  k_layernorm<<<(unsigned)nrows, 256, 0, stream>>>(qz, nsc, nbi, qzn);
  k_tr_pd<<<cdiv(nq, 256), 256, 0, stream>>>(qzn, qznt, nq);
  k_tr_ct<<<cdiv(nq, 256), 256, 0, stream>>>(qzn, qzct, nq);

  // 3. four head projections: per-b GEMM qzn @ W^T -> bf16
  {
    GemmP g{};
    g.A = qzn; g.M = C_ * P_; g.N = D_; g.K = D_;
    g.lda = D_; g.ldb = D_; g.ldc = D_;
    g.asr = (long long)C_ * P_ * D_; g.bsr = (long long)D_ * D_;
    g.bdiv = 1; g.cs0 = (long long)C_ * P_ * D_; g.cs1 = 0;
    dim3 grid(cdiv(D_, 64), cdiv(C_ * P_, 64), B_);
    const __bf16* wl[4] = { wtu, wtv, wcu, wcv };
    for (int j = 0; j < 4; ++j) {
      g.B = wl[j];
      g.outB = proj + (size_t)j * nq;
      k_gemm<EPI_STORE_BF16><<<grid, 128, 0, stream>>>(g);
    }
  }

  // 4. attention probability matrices (head-averaged)
  k_attn_time<<<dim3(16, B_ * C_), 128, 0, stream>>>(proj, proj + nq, attnT);
  k_attn_channel<<<B_ * P_, 32, 0, stream>>>(proj + 2 * nq, proj + 3 * nq, attnC);

  // 5. m_t = attnT @ qzn  (writes msum)
  {
    GemmP g{};
    g.A = attnT; g.B = qznt; g.M = P_; g.N = D_; g.K = P_;
    g.lda = P_; g.ldb = P_; g.ldc = D_;
    g.asr = (long long)P_ * P_; g.bsr = (long long)D_ * P_;
    g.bdiv = 1; g.cs0 = (long long)P_ * D_; g.cs1 = 0;
    g.outF = msum;
    k_gemm<EPI_STORE_F32><<<dim3(4, 4, B_ * C_), 128, 0, stream>>>(g);
  }

  // 6. m_c accumulated into msum (scattered c-rows via ldc = P*D)
  {
    GemmP g{};
    g.A = attnC; g.B = qzct; g.M = C_; g.N = D_; g.K = C_;
    g.lda = C_; g.ldb = C_; g.ldc = P_ * D_;
    g.asr = (long long)C_ * C_; g.bsr = (long long)D_ * C_;
    g.bdiv = P_; g.cs0 = (long long)C_ * P_ * D_; g.cs1 = D_;
    g.outF = msum;
    k_gemm<EPI_ACC_F32><<<dim3(4, 1, B_ * P_), 128, 0, stream>>>(g);
  }

  // 7. topic GEMM1: qg = relu(qzn @ topic^T), rowsum
  {
    GemmP g{};
    g.A = qzn; g.B = topic_bf; g.M = C_ * P_; g.N = G_; g.K = D_;
    g.lda = D_; g.ldb = D_; g.ldc = G_;
    g.asr = (long long)C_ * P_ * D_; g.bsr = (long long)G_ * D_;
    g.bdiv = 1; g.cs0 = (long long)C_ * P_ * G_; g.cs1 = 0;
    g.outB = qg; g.rowsum = rowsum;
    k_gemm<EPI_RELU_RS><<<dim3(8, 128, B_), 128, 0, stream>>>(g);
  }

  // 8. topic GEMM2: msum += (qg @ topic) / max(rowsum, 1e-6)
  {
    GemmP g{};
    g.A = qg; g.B = topicT; g.M = C_ * P_; g.N = D_; g.K = G_;
    g.lda = G_; g.ldb = G_; g.ldc = D_;
    g.asr = (long long)C_ * P_ * G_; g.bsr = (long long)D_ * G_;
    g.bdiv = 1; g.cs0 = (long long)C_ * P_ * D_; g.cs1 = 0;
    g.outF = msum; g.rowsum = rowsum;
    k_gemm<EPI_TOPIC2><<<dim3(4, 128, B_), 128, 0, stream>>>(g);
  }

  // 9. gate GEMM + blended residual -> d_out = qz2
  k_f32_to_bf16<<<cdiv(nq, 256), 256, 0, stream>>>(qz, qzbf, nq);
  {
    GemmP g{};
    g.A = qzbf; g.B = gwb; g.M = (int)nrows; g.N = D_; g.K = D_;
    g.lda = D_; g.ldb = D_; g.ldc = D_;
    g.asr = 0; g.bsr = 0; g.bdiv = 1; g.cs0 = 0; g.cs1 = 0;
    g.bias = gate_b; g.outF = out;
    g.aux0 = qz; g.aux1 = unary; g.aux2 = msum;
    k_gemm<EPI_GATE><<<dim3(4, 1024, 1), 128, 0, stream>>>(g);
  }

  // 10. output MLP: h = LN(qz2); t = gelu(h@w1^T+b1); out += t@w2^T+b2
  k_layernorm<<<(unsigned)nrows, 256, 0, stream>>>(out, nsc, nbi, hbuf);
  {
    GemmP g{};
    g.A = hbuf; g.B = w1b; g.M = (int)nrows; g.N = D_; g.K = D_;
    g.lda = D_; g.ldb = D_; g.ldc = D_;
    g.asr = 0; g.bsr = 0; g.bdiv = 1; g.cs0 = 0; g.cs1 = 0;
    g.bias = b1; g.outB = tbuf;
    k_gemm<EPI_GELU><<<dim3(4, 1024, 1), 128, 0, stream>>>(g);
  }
  {
    GemmP g{};
    g.A = tbuf; g.B = w2b; g.M = (int)nrows; g.N = D_; g.K = D_;
    g.lda = D_; g.ldb = D_; g.ldc = D_;
    g.asr = 0; g.bsr = 0; g.bdiv = 1; g.cs0 = 0; g.cs1 = 0;
    g.bias = b2; g.outF = out;
    k_gemm<EPI_ADDOUT><<<dim3(4, 1024, 1), 128, 0, stream>>>(g);
  }
}
